// TRGTMeanRelationBlock_79860621902502
// MI455X (gfx1250) — compile-verified, matrix operand order checked by reading, not run on hardware
//
#include <hip/hip_runtime.h>
#include <math.h>

#define H 128
#define EDIM 64
#define NN 50000
#define NE 800000

typedef __attribute__((ext_vector_type(16))) __bf16        v16bf;
typedef __attribute__((ext_vector_type(8)))  float         v8f;
typedef __attribute__((ext_vector_type(4)))  unsigned int  u32x4;

union F16F {
    v16bf bf;
    u32x4 q[2];
};

__device__ __forceinline__ unsigned short f2bf(float x) {
    unsigned u = __float_as_uint(x);
    u += 0x7FFFu + ((u >> 16) & 1u);
    return (unsigned short)(u >> 16);
}
__device__ __forceinline__ float bf2f(unsigned short h) {
    return __uint_as_float(((unsigned)h) << 16);
}
// Branch-free exact-GELU: erf via Abramowitz-Stegun 7.1.26 (|err|<1.5e-7).
__device__ __forceinline__ float gelu_exact(float x) {
    float z  = x * 0.70710678118654752f;
    float az = fabsf(z);
    float t  = __builtin_amdgcn_rcpf(1.0f + 0.3275911f * az);
    float p  = t * (0.254829592f +
               t * (-0.284496736f +
               t * (1.421413741f +
               t * (-1.453152027f +
               t * 1.061405429f))));
    float e   = __expf(-az * az);
    float erfv = copysignf(1.0f - p * e, z);
    return 0.5f * x * (1.0f + erfv);
}
__device__ __forceinline__ float sigm(float x) {
    return 1.0f / (1.0f + __expf(-x));
}

// Load a 16x32 bf16 WMMA operand fragment from a row-major LDS array.
// Lanes 0-15: row=lane, K = kbase+[0,8) and kbase+[16,24);
// lanes 16-31: row=lane-16, K = kbase+[8,16) and kbase+[24,32).
// Pass row = lane&15 for A; for B pass row = output column (array is W^T).
__device__ __forceinline__ v16bf ldsfrag(const unsigned short* base, int row,
                                         int stride, int kbase, int lane) {
    int hl = lane >> 4;
    const unsigned short* p = base + row * stride + kbase + hl * 8;
    F16F f;
    f.q[0] = *(const u32x4*)(p);
    f.q[1] = *(const u32x4*)(p + 16);
    return f.bf;
}

#define WMMA_BF16(A, B, C) \
    __builtin_amdgcn_wmma_f32_16x16x32_bf16(false, (A), false, (B), (short)0, (C), false, false)

// ---------------------------------------------------------------- LayerNorm 1
// Writes h as bf16 (packed) so edge/node gathers move half the bytes.
__global__ void k_ln1(const float* __restrict__ x, const float* __restrict__ g,
                      const float* __restrict__ b, unsigned short* __restrict__ h) {
    int wid = threadIdx.x >> 5, lane = threadIdx.x & 31;
    int row = blockIdx.x * 8 + wid;
    if (row >= NN) return;
    const float* xr = x + (size_t)row * H;
    float4 v = *(const float4*)(xr + lane * 4);
    float s = v.x + v.y + v.z + v.w;
    float q = v.x * v.x + v.y * v.y + v.z * v.z + v.w * v.w;
    for (int off = 16; off > 0; off >>= 1) {
        s += __shfl_xor(s, off, 32);
        q += __shfl_xor(q, off, 32);
    }
    float mean = s * (1.0f / H);
    float var  = q * (1.0f / H) - mean * mean;
    float rstd = rsqrtf(var + 1e-5f);
    int c = lane * 4;
    float o0 = (v.x - mean) * rstd * g[c + 0] + b[c + 0];
    float o1 = (v.y - mean) * rstd * g[c + 1] + b[c + 1];
    float o2 = (v.z - mean) * rstd * g[c + 2] + b[c + 2];
    float o3 = (v.w - mean) * rstd * g[c + 3] + b[c + 3];
    unsigned p0 = (unsigned)f2bf(o0) | ((unsigned)f2bf(o1) << 16);
    unsigned p1 = (unsigned)f2bf(o2) | ((unsigned)f2bf(o3) << 16);
    unsigned long long pk = ((unsigned long long)p1 << 32) | (unsigned long long)p0;
    *(unsigned long long*)(h + (size_t)row * H + c) = pk;
}

// ---------------------------------------------------------------- degree
__global__ void k_deg(const int* __restrict__ dst, float* __restrict__ deg) {
    int e = blockIdx.x * blockDim.x + threadIdx.x;
    if (e < NE) atomicAdd(&deg[dst[e]], 1.0f);
}

// ---------------------------------------------------------------- edge MLPs
// LDS layout (bytes):
//   [0,81920)        Wgate1^T  bf16 [128][320]
//   [81920,131072)   WmsgW1^T  bf16 [128][192]
//   [131072,163840)  WmsgW2^T  bf16 [128][128]
//   [163840,164352)  gateW2    f32  [128]
//   [164352,164864)  msgb1     f32  [128]
//   [164864,165376)  msgb2     f32  [128]
//   [165376,165888)  gateb1    f32  [128]
//   [165888,176128)  aStage    bf16 [16][320]   (hd|hs|e per edge row)
//   [176128,180224)  hStage    bf16 [16][128]
//   [180224,180288)  gateval   f32  [16]
//   [180288,180352)  dstIdx    i32  [16]
//   [180352,181376)  sRed      f32  [256]
#define EDGE_LDS 181376

__global__ void __launch_bounds__(256) k_edge(
    const unsigned short* __restrict__ h, const float* __restrict__ edge_emb,
    const float* __restrict__ time_w, const float* __restrict__ node_scale,
    const float* __restrict__ msgW1, const float* __restrict__ msgb1,
    const float* __restrict__ msgW2, const float* __restrict__ msgb2,
    const float* __restrict__ gateW1, const float* __restrict__ gateb1,
    const float* __restrict__ gateW2, const float* __restrict__ gateb2,
    const int* __restrict__ edge_src, const int* __restrict__ edge_dst,
    float* __restrict__ agg, float* __restrict__ edge_out) {

    extern __shared__ char smem[];
    unsigned short* sWg1 = (unsigned short*)(smem);
    unsigned short* sWm1 = (unsigned short*)(smem + 81920);
    unsigned short* sWm2 = (unsigned short*)(smem + 131072);
    float* sGw2 = (float*)(smem + 163840);
    float* sB1m = (float*)(smem + 164352);
    float* sB2m = (float*)(smem + 164864);
    float* sB1g = (float*)(smem + 165376);
    unsigned short* aStage = (unsigned short*)(smem + 165888);
    unsigned short* hStage = (unsigned short*)(smem + 176128);
    float* gateval = (float*)(smem + 180224);
    int*   dstIdx  = (int*)(smem + 180288);
    float* sRed    = (float*)(smem + 180352);

    const int tid = threadIdx.x, lane = tid & 31, wid = tid >> 5;
    const int nlo = lane & 15, hl = lane >> 4;
    const int col = wid * 16 + nlo;

    // ---- transpose-convert weights f32 -> bf16 LDS (W^T[out][in]) ----
    for (int idx = tid; idx < 320 * 128; idx += 256) {
        int o = idx & 127, i = idx >> 7;
        sWg1[o * 320 + i] = f2bf(gateW1[idx]);
    }
    for (int idx = tid; idx < 192 * 128; idx += 256) {
        int o = idx & 127, i = idx >> 7;
        sWm1[o * 192 + i] = f2bf(msgW1[idx]);
    }
    for (int idx = tid; idx < 128 * 128; idx += 256) {
        int o = idx & 127, i = idx >> 7;
        sWm2[o * 128 + i] = f2bf(msgW2[idx]);
    }
    if (tid < 128) {
        sGw2[tid] = gateW2[tid];
        sB1m[tid] = msgb1[tid];
        sB2m[tid] = msgb2[tid];
        sB1g[tid] = gateb1[tid];
    }
    __syncthreads();

    const unsigned lds_aStage = (unsigned)(size_t)(void*)aStage;
    const unsigned long long hbase = (unsigned long long)(size_t)(const void*)h;

    const int ntiles = NE / 16;
    for (int t = blockIdx.x; t < ntiles; t += gridDim.x) {
        const int ebase = t * 16;

        // ---- async gather hd|hs (bf16, 16B chunks) straight into LDS ----
        // 16 rows x 32 chunks: chunk<16 -> hd cols [chunk*8,..), else hs.
        // Uniform 2 iterations per thread -> EXEC stays full.
#pragma unroll
        for (int c0 = 0; c0 < 512; c0 += 256) {
            int c = c0 + tid;
            int row = c >> 5, chunk = c & 31;
            int e = ebase + row;
            int node = (chunk < 16) ? edge_dst[e] : edge_src[e];
            unsigned goff = ((unsigned)node * 128u + (unsigned)(chunk & 15) * 8u) * 2u;
            unsigned loff = lds_aStage + (unsigned)(row * 320 + chunk * 8) * 2u;
            asm volatile("global_load_async_to_lds_b128 %0, %1, %2 offset:0"
                         :: "v"(loff), "v"(goff), "s"(hbase) : "memory");
        }
        // ---- edge_emb: f32 -> bf16 into cols [256,320) ----
        if (tid < 128) {
            int row = tid >> 3, chunk = tid & 7;
            const float* p = edge_emb + (size_t)(ebase + row) * EDIM + chunk * 8;
            unsigned short* q = aStage + row * 320 + 256 + chunk * 8;
#pragma unroll
            for (int j = 0; j < 8; ++j) q[j] = f2bf(p[j]);
        }
        asm volatile("s_wait_asynccnt 0" ::: "memory");
        __syncthreads();

        // ---- msg GEMM1: (16x192 = hs|e) @ msgW1 -> GELU -> hStage ----
        v8f hh;
        {
            float bb = sB1m[col];
#pragma unroll
            for (int r = 0; r < 8; ++r) hh[r] = bb;
        }
#pragma unroll
        for (int k = 0; k < 192; k += 32) {
            v16bf a = ldsfrag(aStage, nlo, 320, 128 + k, lane);
            v16bf b = ldsfrag(sWm1, col, 192, k, lane);
            hh = WMMA_BF16(a, b, hh);
        }
#pragma unroll
        for (int r = 0; r < 8; ++r) {
            float gv = gelu_exact(hh[r]);
            hStage[(r + 8 * hl) * 128 + col] = f2bf(gv);
        }
        __syncthreads();

        // ---- msg GEMM2: hidden(16x128) @ msgW2 -> accM ----
        v8f accM;
        {
            float bb = sB2m[col];
#pragma unroll
            for (int r = 0; r < 8; ++r) accM[r] = bb;
        }
#pragma unroll
        for (int k = 0; k < 128; k += 32) {
            v16bf a = ldsfrag(hStage, nlo, 128, k, lane);
            v16bf b = ldsfrag(sWm2, col, 128, k, lane);
            accM = WMMA_BF16(a, b, accM);
        }
        __syncthreads();  // all reads of hStage done before gate overwrites it

        // ---- gate GEMM1: (16x320 = hd|hs|e) @ gateW1 -> GELU -> hStage ----
        v8f gg;
        {
            float bb = sB1g[col];
#pragma unroll
            for (int r = 0; r < 8; ++r) gg[r] = bb;
        }
#pragma unroll
        for (int k = 0; k < 320; k += 32) {
            v16bf a = ldsfrag(aStage, nlo, 320, k, lane);
            v16bf b = ldsfrag(sWg1, col, 320, k, lane);
            gg = WMMA_BF16(a, b, gg);
        }
#pragma unroll
        for (int r = 0; r < 8; ++r) {
            float gv = gelu_exact(gg[r]);
            hStage[(r + 8 * hl) * 128 + col] = f2bf(gv);
        }
        __syncthreads();

        // ---- gate scalar: block-wide partial dot, then 16-way finalize ----
        {
            int row = tid >> 4, seg = tid & 15;
            float s = 0.f;
#pragma unroll
            for (int j = 0; j < 8; ++j) {
                int k = seg * 8 + j;
                s += bf2f(hStage[row * 128 + k]) * sGw2[k];
            }
            sRed[tid] = s;
        }
        __syncthreads();
        if (tid < 16) {
            float d = gateb2[0];
#pragma unroll
            for (int j = 0; j < 16; ++j) d += sRed[tid * 16 + j];
            int e = ebase + tid;
            int ssrc = edge_src[e], dd = edge_dst[e];
            float esc = 0.5f * (node_scale[ssrc] + node_scale[dd]);
            gateval[tid] = sigm(d) * time_w[e] * esc;
            dstIdx[tid] = dd;
        }
        __syncthreads();

        // ---- finalize: edge_repr out + f32 atomic aggregation ----
#pragma unroll
        for (int r = 0; r < 8; ++r) {
            int m = r + 8 * hl;
            float v = accM[r] * gateval[m];
            edge_out[(size_t)(ebase + m) * H + col] = v;
            atomicAdd(&agg[(size_t)dstIdx[m] * H + col], v);
        }
        __syncthreads();
    }
}

// ---------------------------------------------------------------- node update
// LDS: sWs[128][128] bf16 | sWa[128][128] bf16 | sb f32[128] | aH[16][128] | aA[16][128]
#define NODE_LDS 74240

__global__ void __launch_bounds__(256) k_node(
    const float* __restrict__ x, const unsigned short* __restrict__ h,
    const float* __restrict__ agg, const float* __restrict__ deg,
    const float* __restrict__ selfW, const float* __restrict__ selfb,
    const float* __restrict__ aggW, const float* __restrict__ aggb,
    float* __restrict__ mid) {

    extern __shared__ char smem[];
    unsigned short* sWs = (unsigned short*)(smem);
    unsigned short* sWa = (unsigned short*)(smem + 32768);
    float* sb = (float*)(smem + 65536);
    unsigned short* aH = (unsigned short*)(smem + 66048);
    unsigned short* aA = (unsigned short*)(smem + 70144);

    const int tid = threadIdx.x, lane = tid & 31, wid = tid >> 5;
    const int nlo = lane & 15, hl = lane >> 4;
    const int col = wid * 16 + nlo;

    for (int idx = tid; idx < 128 * 128; idx += 256) {
        int o = idx & 127, i = idx >> 7;
        sWs[o * 128 + i] = f2bf(selfW[idx]);
        sWa[o * 128 + i] = f2bf(aggW[idx]);
    }
    if (tid < 128) sb[tid] = selfb[tid] + aggb[tid];
    __syncthreads();

    const int ntiles = NN / 16;
    for (int t = blockIdx.x; t < ntiles; t += gridDim.x) {
        const int nb = t * 16;
        {   // 256 chunks of 8 elements: exactly one per thread
            int row = tid >> 4, chunk = tid & 15;
            int node = nb + row;
            // h is already bf16: straight 16B copy
            *(u32x4*)(aH + row * 128 + chunk * 8) =
                *(const u32x4*)(h + (size_t)node * H + chunk * 8);
            float dg = fmaxf(deg[node], 1.0f);
            const float* ap = agg + (size_t)node * H + chunk * 8;
#pragma unroll
            for (int j = 0; j < 8; ++j)
                aA[row * 128 + chunk * 8 + j] = f2bf(ap[j] / dg);
        }
        __syncthreads();

        v8f acc;
        {
            float bb = sb[col];
#pragma unroll
            for (int r = 0; r < 8; ++r) acc[r] = bb;
        }
#pragma unroll
        for (int k = 0; k < 128; k += 32) {
            acc = WMMA_BF16(ldsfrag(aH, nlo, 128, k, lane),
                            ldsfrag(sWs, col, 128, k, lane), acc);
            acc = WMMA_BF16(ldsfrag(aA, nlo, 128, k, lane),
                            ldsfrag(sWa, col, 128, k, lane), acc);
        }
#pragma unroll
        for (int r = 0; r < 8; ++r) {
            int m = r + 8 * hl;
            size_t off = (size_t)(nb + m) * H + col;
            mid[off] = x[off] + acc[r];
        }
        __syncthreads();
    }
}

// ---------------------------------------------------------------- LN2 + FFN
// LDS: sW1[256][128] | sW2[128][256] | sb1[256] | sb2[128] | aS[16][128] bf16 |
//      hS[16][256] bf16 | mS[16][128] f32 | redS[256] | redQ[256] | stats[32]
#define FFN_LDS 155264

__global__ void __launch_bounds__(256) k_ffn(
    const float* __restrict__ mid, const float* __restrict__ g2,
    const float* __restrict__ b2, const float* __restrict__ ffnW1,
    const float* __restrict__ ffnb1, const float* __restrict__ ffnW2,
    const float* __restrict__ ffnb2, float* __restrict__ out) {

    extern __shared__ char smem[];
    unsigned short* sW1 = (unsigned short*)(smem);
    unsigned short* sW2 = (unsigned short*)(smem + 65536);
    float* sb1 = (float*)(smem + 131072);
    float* sb2 = (float*)(smem + 132096);
    unsigned short* aS = (unsigned short*)(smem + 132608);
    unsigned short* hS = (unsigned short*)(smem + 136704);
    float* mS = (float*)(smem + 144896);
    float* redS = (float*)(smem + 153088);
    float* redQ = (float*)(smem + 154112);
    float* stats = (float*)(smem + 155136);

    const int tid = threadIdx.x, lane = tid & 31, wid = tid >> 5;
    const int nlo = lane & 15, hl = lane >> 4;

    for (int idx = tid; idx < 128 * 256; idx += 256) {
        int o = idx & 255, i = idx >> 8;  // ffnW1 is [128][256]
        sW1[o * 128 + i] = f2bf(ffnW1[idx]);
    }
    for (int idx = tid; idx < 256 * 128; idx += 256) {
        int o = idx & 127, i = idx >> 7;  // ffnW2 is [256][128]
        sW2[o * 256 + i] = f2bf(ffnW2[idx]);
    }
    sb1[tid] = ffnb1[tid];
    if (tid < 128) sb2[tid] = ffnb2[tid];
    __syncthreads();

    const int ntiles = NN / 16;
    for (int t = blockIdx.x; t < ntiles; t += gridDim.x) {
        const int nb = t * 16;
        for (int c = tid; c < 2048; c += 256)
            mS[c] = mid[(size_t)nb * H + c];
        __syncthreads();
        {   // per-row partial LN stats
            int row = tid >> 4, s8 = (tid & 15) * 8;
            float s = 0.f, q = 0.f;
#pragma unroll
            for (int j = 0; j < 8; ++j) {
                float v = mS[row * 128 + s8 + j];
                s += v; q += v * v;
            }
            redS[tid] = s; redQ[tid] = q;
        }
        __syncthreads();
        if (tid < 16) {
            float s = 0.f, q = 0.f;
#pragma unroll
            for (int j = 0; j < 16; ++j) { s += redS[tid * 16 + j]; q += redQ[tid * 16 + j]; }
            float m = s / 128.0f, v = q / 128.0f - m * m;
            stats[2 * tid] = m;
            stats[2 * tid + 1] = rsqrtf(v + 1e-5f);
        }
        __syncthreads();
        for (int c = tid; c < 2048; c += 256) {
            int row = c >> 7, ccol = c & 127;
            float v = (mS[c] - stats[2 * row]) * stats[2 * row + 1] * g2[ccol] + b2[ccol];
            aS[c] = f2bf(v);
        }
        __syncthreads();

        // GEMM1: 16x128 @ 128x256 ; each wave does N-tiles wid and wid+8
        int colA = wid * 16 + nlo, colB = (wid + 8) * 16 + nlo;
        v8f ha, hb;
#pragma unroll
        for (int r = 0; r < 8; ++r) { ha[r] = sb1[colA]; hb[r] = sb1[colB]; }
#pragma unroll
        for (int k = 0; k < 128; k += 32) {
            v16bf a  = ldsfrag(aS, nlo, 128, k, lane);
            v16bf wA = ldsfrag(sW1, colA, 128, k, lane);
            v16bf wB = ldsfrag(sW1, colB, 128, k, lane);
            ha = WMMA_BF16(a, wA, ha);
            hb = WMMA_BF16(a, wB, hb);
        }
#pragma unroll
        for (int r = 0; r < 8; ++r) {
            int m = r + 8 * hl;
            hS[m * 256 + colA] = f2bf(gelu_exact(ha[r]));
            hS[m * 256 + colB] = f2bf(gelu_exact(hb[r]));
        }
        __syncthreads();

        // GEMM2: 16x256 @ 256x128 + residual
        int col = wid * 16 + nlo;
        v8f acc;
#pragma unroll
        for (int r = 0; r < 8; ++r) acc[r] = sb2[col];
#pragma unroll
        for (int k = 0; k < 256; k += 32) {
            acc = WMMA_BF16(ldsfrag(hS, nlo, 256, k, lane),
                            ldsfrag(sW2, col, 256, k, lane), acc);
        }
#pragma unroll
        for (int r = 0; r < 8; ++r) {
            int m = r + 8 * hl;
            out[(size_t)(nb + m) * H + col] = mS[m * 128 + col] + acc[r];
        }
        __syncthreads();
    }
}

// ---------------------------------------------------------------- launcher
extern "C" void kernel_launch(void* const* d_in, const int* in_sizes, int n_in,
                              void* d_out, int out_size, void* d_ws, size_t ws_size,
                              hipStream_t stream) {
    (void)in_sizes; (void)n_in; (void)out_size; (void)ws_size;

    const float* x          = (const float*)d_in[0];
    const float* edge_emb   = (const float*)d_in[1];
    const float* time_w     = (const float*)d_in[2];
    const float* node_scale = (const float*)d_in[3];
    const float* ln1_g = (const float*)d_in[4];
    const float* ln1_b = (const float*)d_in[5];
    const float* ln2_g = (const float*)d_in[6];
    const float* ln2_b = (const float*)d_in[7];
    const float* selfW = (const float*)d_in[8];
    const float* selfb = (const float*)d_in[9];
    const float* msgW1 = (const float*)d_in[10];
    const float* msgb1 = (const float*)d_in[11];
    const float* msgW2 = (const float*)d_in[12];
    const float* msgb2 = (const float*)d_in[13];
    const float* gateW1 = (const float*)d_in[14];
    const float* gateb1 = (const float*)d_in[15];
    const float* gateW2 = (const float*)d_in[16];
    const float* gateb2 = (const float*)d_in[17];
    const float* aggW = (const float*)d_in[18];
    const float* aggb = (const float*)d_in[19];
    const float* ffnW1 = (const float*)d_in[20];
    const float* ffnb1 = (const float*)d_in[21];
    const float* ffnW2 = (const float*)d_in[22];
    const float* ffnb2 = (const float*)d_in[23];
    const int* edge_src = (const int*)d_in[24];
    const int* edge_dst = (const int*)d_in[25];

    float* out_nodes = (float*)d_out;
    float* edge_out  = out_nodes + (size_t)NN * H;

    unsigned short* hb = (unsigned short*)d_ws;                    // NN*H bf16
    float* mid = (float*)((char*)d_ws + (size_t)NN * H * 2);       // NN*H f32
    float* agg = mid + (size_t)NN * H;                             // NN*H f32
    float* deg = agg + (size_t)NN * H;                             // NN f32

    hipMemsetAsync(agg, 0, ((size_t)NN * H + NN) * sizeof(float), stream);

    k_ln1<<<NN / 8, 256, 0, stream>>>(x, ln1_g, ln1_b, hb);
    k_deg<<<(NE + 255) / 256, 256, 0, stream>>>(edge_dst, deg);

    hipFuncSetAttribute((const void*)k_edge,
                        hipFuncAttributeMaxDynamicSharedMemorySize, EDGE_LDS);
    k_edge<<<1024, 256, EDGE_LDS, stream>>>(
        hb, edge_emb, time_w, node_scale, msgW1, msgb1, msgW2, msgb2,
        gateW1, gateb1, gateW2, gateb2, edge_src, edge_dst, agg, edge_out);

    hipFuncSetAttribute((const void*)k_node,
                        hipFuncAttributeMaxDynamicSharedMemorySize, NODE_LDS);
    k_node<<<320, 256, NODE_LDS, stream>>>(
        x, hb, agg, deg, selfW, selfb, aggW, aggb, mid);

    hipFuncSetAttribute((const void*)k_ffn,
                        hipFuncAttributeMaxDynamicSharedMemorySize, FFN_LDS);
    k_ffn<<<320, 256, FFN_LDS, stream>>>(
        mid, ln2_g, ln2_b, ffnW1, ffnb1, ffnW2, ffnb2, out_nodes);
}
